// LSTM_876173328464
// MI455X (gfx1250) — compile-verified, matrix operand order checked by reading, not run on hardware
//
#include <hip/hip_runtime.h>

#define B_TOTAL 8192
#define T_STEPS 256
#define IN_F    16
#define HID     32
#define L2E     1.4426950408889634f

typedef __attribute__((ext_vector_type(16))) _Float16 v16h;
typedef __attribute__((ext_vector_type(8)))  _Float16 v8h;
typedef __attribute__((ext_vector_type(8)))  float    v8f;

#if __has_builtin(__builtin_amdgcn_exp2f)
static __device__ __forceinline__ float fexp2(float x) { return __builtin_amdgcn_exp2f(x); }
#else
static __device__ __forceinline__ float fexp2(float x) { return __expf(x * 0.6931471805599453f); }
#endif

// sigmoid(g + b) with nb = -L2E*b precomputed: one v_fma + v_exp + v_add + v_rcp
static __device__ __forceinline__ float sigm_b(float g, float nb) {
  return __builtin_amdgcn_rcpf(1.0f + fexp2(__builtin_fmaf(g, -L2E, nb)));
}
// tanh(g + b) with nb = -2*L2E*b precomputed
static __device__ __forceinline__ float tanh_b(float g, float nb) {
  return 2.0f * __builtin_amdgcn_rcpf(1.0f + fexp2(__builtin_fmaf(g, -2.0f * L2E, nb))) - 1.0f;
}
// tanh(c), no bias
static __device__ __forceinline__ float tanh_nb(float c) {
  return 2.0f * __builtin_amdgcn_rcpf(1.0f + fexp2(c * (-2.0f * L2E))) - 1.0f;
}

// Load a 16x32 f16 A-fragment from a row-major [16 x 32] LDS tile.
// A layout (16-bit A 16x32): lanes 0-15: M=lane, K={0..7,16..23};
// lanes 16-31: M=lane-16, K={8..15,24..31}.
static __device__ __forceinline__ v16h afrag_from_lds(const _Float16* h, int lane) {
  const int m  = lane & 15;
  const int kb = (lane < 16) ? 0 : 8;
  const _Float16* p = h + m * HID + kb;          // 16B aligned
  v8h lo = *(const v8h*)(p);                     // K = kb..kb+7
  v8h hi = *(const v8h*)(p + 16);                // K = 16+kb..16+kb+7
  return __builtin_shufflevector(lo, hi, 0,1,2,3,4,5,6,7,8,9,10,11,12,13,14,15);
}

// Build a 32x16 f16 B-fragment (K x N) from row-major W[4H][ld]: B[k][n] = W[tile*16+n][k].
static __device__ __forceinline__ v16h bfrag_from_w(const float* W, int ld, int tile,
                                                    bool padk, int lane) {
  const int n  = tile * 16 + (lane & 15);
  const int kb = (lane < 16) ? 0 : 8;
  const float* row = W + n * ld;
  v16h b;
#pragma unroll
  for (int i = 0; i < 8; ++i) b[i] = (_Float16)row[kb + i];
#pragma unroll
  for (int i = 0; i < 8; ++i) b[8 + i] = padk ? (_Float16)0.0f : (_Float16)row[16 + kb + i];
  return b;
}

// Half-cell: this wave owns gate-column half `w` (cols w*16 .. w*16+15) = tiles {w,2+w,4+w,6+w}.
// gates = in*Wi^T + h*Wh^T (bias folded into activations); c/h update; h half written to LDS.
static __device__ __forceinline__ void lstm_half_cell(const v16h in, const v16h hr,
                                                      const v16h* Wi, const v16h* Wh,
                                                      const float* nb,  // scaled neg bias [4 gates]
                                                      v8f& c, _Float16* hout,
                                                      int lane, int w) {
  v8f g[4];
#pragma unroll
  for (int gt = 0; gt < 4; ++gt) {               // i, f, g, o
    v8f acc = {};                                // lowers to inline-0 C operand
    acc = __builtin_amdgcn_wmma_f32_16x16x32_f16(false, in, false, Wi[gt], (short)0, acc, false, false);
    acc = __builtin_amdgcn_wmma_f32_16x16x32_f16(false, hr, false, Wh[gt], (short)0, acc, false, false);
    g[gt] = acc;
  }
  const int col = w * 16 + (lane & 15);
  const int mb  = (lane < 16) ? 0 : 8;
#pragma unroll
  for (int r = 0; r < 8; ++r) {                  // batch rows mb..mb+7
    float iv = sigm_b(g[0][r], nb[0]);
    float fv = sigm_b(g[1][r], nb[1]);
    float gv = tanh_b(g[2][r], nb[2]);
    float ov = sigm_b(g[3][r], nb[3]);
    float cv = __builtin_fmaf(fv, c[r], iv * gv);
    c[r] = cv;
    float hv = ov * tanh_nb(cv);
    hout[(mb + r) * HID + col] = (_Float16)hv;
  }
}

extern "C" __global__ __launch_bounds__(64)
void lstm2_wmma_fused(const float* __restrict__ x,
                      const float* __restrict__ Wih0, const float* __restrict__ Whh0,
                      const float* __restrict__ bih0, const float* __restrict__ bhh0,
                      const float* __restrict__ Wih1, const float* __restrict__ Whh1,
                      const float* __restrict__ bih1, const float* __restrict__ bhh1,
                      const float* __restrict__ Wfc,  const float* __restrict__ bfc,
                      float* __restrict__ out) {
  __shared__ _Float16 hA[16 * HID];   // layer-0 h staging (16x32 f16, row-major)
  __shared__ _Float16 hB[16 * HID];   // layer-1 h staging
  const int lane = threadIdx.x & 31;
  const int w    = threadIdx.x >> 5;  // 0/1: which 16-column gate half this wave owns
  const int b0   = blockIdx.x * 16;

  // ---- per-wave weight fragments: 4 gates x (Wih, Whh) x 2 layers = 16 frags (128 VGPRs)
  v16h wi0[4], wh0[4], wi1[4], wh1[4];
#pragma unroll
  for (int gt = 0; gt < 4; ++gt) {
    const int tile = gt * 2 + w;
    wi0[gt] = bfrag_from_w(Wih0, IN_F, tile, true,  lane);   // K=16 padded to 32
    wh0[gt] = bfrag_from_w(Whh0, HID,  tile, false, lane);
    wi1[gt] = bfrag_from_w(Wih1, HID,  tile, false, lane);
    wh1[gt] = bfrag_from_w(Whh1, HID,  tile, false, lane);
  }
  // ---- scaled negative biases (consumed by one v_fma inside the activation)
  float nb0[4], nb1[4];
#pragma unroll
  for (int gt = 0; gt < 4; ++gt) {
    const int col   = (gt * 2 + w) * 16 + (lane & 15);
    const float scl = (gt == 2) ? (-2.0f * L2E) : (-L2E);    // gate 2 = tanh gate
    nb0[gt] = scl * (bih0[col] + bhh0[col]);
    nb1[gt] = scl * (bih1[col] + bhh1[col]);
  }
  // ---- FC weights into column 0 of a B fragment
  v16h wfcf;
  {
    const int  kb = (lane < 16) ? 0 : 8;
    const bool n0 = (lane & 15) == 0;
#pragma unroll
    for (int i = 0; i < 8; ++i) wfcf[i]     = n0 ? (_Float16)Wfc[kb + i]      : (_Float16)0.0f;
#pragma unroll
    for (int i = 0; i < 8; ++i) wfcf[8 + i] = n0 ? (_Float16)Wfc[16 + kb + i] : (_Float16)0.0f;
  }

  // ---- state: this wave's c half in D layout; h as full A fragments (h_0 = c_0 = 0)
  v8f c0 = {}, c1 = {};
  v16h hAf, hBf;
#pragma unroll
  for (int i = 0; i < 16; ++i) { hAf[i] = (_Float16)0.0f; hBf[i] = (_Float16)0.0f; }

  const int m_row = lane & 15;
  const int kb_x  = (lane < 16) ? 0 : 8;
  const float* xrow = x + (size_t)(b0 + m_row) * T_STEPS * IN_F + kb_x;

#pragma unroll 1
  for (int t = 0; t < T_STEPS; ++t) {
    // x fragment: K 0..15 = features, K 16..31 = 0 (zeros make Wih0 pad rows don't-care)
    v16h xf;
    const float* p = xrow + (size_t)t * IN_F;
#pragma unroll
    for (int i = 0; i < 8; ++i) xf[i] = (_Float16)p[i];
#pragma unroll
    for (int i = 0; i < 8; ++i) xf[8 + i] = (_Float16)0.0f;
    {
      const int tn = (t + 4 < T_STEPS) ? t + 4 : (T_STEPS - 1);
      __builtin_prefetch(xrow + (size_t)tn * IN_F, 0, 3);  // global_prefetch_b8, keep near
    }

    // layer 0: h1_t = cell(x_t, h1_{t-1}); each wave computes its 16-column half
    lstm_half_cell(xf, hAf, wi0, wh0, nb0, c0, hA, lane, w);
    __syncthreads();                     // both halves of h1_t in LDS
    hAf = afrag_from_lds(hA, lane);      // full h1_t: layer-1 input + recurrent for t+1

    // layer 1: h2_t = cell(h1_t, h2_{t-1})
    lstm_half_cell(hAf, hBf, wi1, wh1, nb1, c1, hB, lane, w);
    __syncthreads();
    hBf = afrag_from_lds(hB, lane);      // full h2_t
  }

  // ---- final FC (wave 0 only): one WMMA, result in column 0 of the D tile
  if (w == 0) {
    v8f od;
    const float fb = ((lane & 15) == 0) ? bfc[0] : 0.0f;
#pragma unroll
    for (int i = 0; i < 8; ++i) od[i] = fb;
    od = __builtin_amdgcn_wmma_f32_16x16x32_f16(false, hBf, false, wfcf, (short)0, od, false, false);
    if ((lane & 15) == 0) {
      const int mb = (lane < 16) ? 0 : 8;
#pragma unroll
      for (int r = 0; r < 8; ++r) out[b0 + mb + r] = od[r];
    }
  }
}

extern "C" void kernel_launch(void* const* d_in, const int* in_sizes, int n_in,
                              void* d_out, int out_size, void* d_ws, size_t ws_size,
                              hipStream_t stream) {
  const float* x    = (const float*)d_in[0];
  const float* Wih0 = (const float*)d_in[1];
  const float* Whh0 = (const float*)d_in[2];
  const float* bih0 = (const float*)d_in[3];
  const float* bhh0 = (const float*)d_in[4];
  const float* Wih1 = (const float*)d_in[5];
  const float* Whh1 = (const float*)d_in[6];
  const float* bih1 = (const float*)d_in[7];
  const float* bhh1 = (const float*)d_in[8];
  const float* Wfc  = (const float*)d_in[9];
  const float* bfc  = (const float*)d_in[10];
  float* out = (float*)d_out;

  dim3 grid(B_TOTAL / 16);   // 512 blocks, one 16-batch tile each
  dim3 block(64);            // 2 wave32s: gate-column halves
  lstm2_wmma_fused<<<grid, block, 0, stream>>>(x, Wih0, Whh0, bih0, bhh0,
                                               Wih1, Whh1, bih1, bhh1,
                                               Wfc, bfc, out);
}
